// MultiHeadSelfAttention_1030792151557
// MI455X (gfx1250) — matrix operand order repaired and checked by reading, not compiled, not run on hardware
//
#include <hip/hip_runtime.h>
#include <hip/hip_bf16.h>

typedef __bf16 bf16_t;
typedef __attribute__((ext_vector_type(16))) __bf16 v16bf;
typedef __attribute__((ext_vector_type(8)))  __bf16 v8bf;
typedef __attribute__((ext_vector_type(4)))  __bf16 v4bf;
typedef __attribute__((ext_vector_type(8)))  float  v8f;
typedef unsigned int v4u __attribute__((ext_vector_type(4)));
typedef int          v4i __attribute__((ext_vector_type(4)));
typedef int          v8i __attribute__((ext_vector_type(8)));

#define DIMC   1024
#define NHEAD  16
#define HDIM   64
#define SEQ    2048
#define BATCH  2
#define TOKENS (BATCH * SEQ)   // 4096
#define QKSCALE 0.125f         // 64^-0.5
#define LDS_LD  72             // 64 elems + 8 elem pad (TDM pad: 128B + 16B)

// ---------------- WMMA fragment helpers (CDNA5 16x16x32 bf16) ----------------
// A-frag (16x32, MxK): lane L: row = L&15, half g = L>>4.
//   VGPR0..3 hold K = g*8 + {0..7}; VGPR4..7 hold K = 16 + g*8 + {0..7}.
__device__ __forceinline__ v16bf load_a_frag(const bf16_t* __restrict__ base,
                                             int ld, int row0, int k0, int lane) {
  const int r = lane & 15, g = lane >> 4;
  const bf16_t* p = base + (size_t)(row0 + r) * ld + k0 + g * 8;
  v8bf lo = *(const v8bf*)(p);
  v8bf hi = *(const v8bf*)(p + 16);
  return __builtin_shufflevector(lo, hi, 0,1,2,3,4,5,6,7,8,9,10,11,12,13,14,15);
}

// B-frag (32x16, KxN): lane L: col n = L&15, half g = L>>4; K = g*8 + {0..7}, 16 + g*8 + {0..7}.
// Source is row-major with ROW index = n, COL index = k (y = A*W^T layout).
__device__ __forceinline__ v16bf load_b_frag(const bf16_t* __restrict__ base,
                                             int ld, int n0, int k0, int lane) {
  const int n = lane & 15, g = lane >> 4;
  const bf16_t* p = base + (size_t)(n0 + n) * ld + k0 + g * 8;
  v8bf lo = *(const v8bf*)(p);
  v8bf hi = *(const v8bf*)(p + 16);
  return __builtin_shufflevector(lo, hi, 0,1,2,3,4,5,6,7,8,9,10,11,12,13,14,15);
}

__device__ __forceinline__ v8f wmma_bf16(v16bf a, v16bf b, v8f c) {
  return __builtin_amdgcn_wmma_f32_16x16x32_bf16(false, a, false, b, (short)0, c,
                                                 false, false);
}

// ---------------- TDM: 2-D tile (64 rows x 64 bf16) global -> LDS -----------
// Pads each 128B row with 16B in LDS -> row stride 144B (LDS_LD elements).
__device__ __forceinline__ void tdm_load_tile(unsigned lds_off,
                                              const bf16_t* gptr,
                                              unsigned long long row_stride_elems) {
  const unsigned long long ga = (unsigned long long)(size_t)gptr;
  v4u g0;
  g0[0] = 1u;                                              // count=1, user desc
  g0[1] = lds_off;                                         // lds_addr (bytes)
  g0[2] = (unsigned)ga;                                    // global_addr[31:0]
  g0[3] = (unsigned)((ga >> 32) & 0x01FFFFFFu) | (2u << 30); // addr[56:32]|type=2
  v8i g1;
  const unsigned td0 = 1u << 20, td1 = 1u << 20;           // huge dims: no OOB
  g1[0] = (int)((1u << 16)       // data_size = 2 bytes
              | (1u << 20)       // pad_enable
              | (4u << 22)       // pad_interval: 32 DWORDs (=128B row)
              | (3u << 25));     // pad_amount:   4 DWORDs (=16B)
  g1[1] = (int)((td0 & 0xFFFFu) << 16);                    // tensor_dim0 lo
  g1[2] = (int)(((td0 >> 16) & 0xFFFFu) | ((td1 & 0xFFFFu) << 16));
  g1[3] = (int)(((td1 >> 16) & 0xFFFFu) | (64u << 16));    // tile_dim0 = 64
  g1[4] = (int)(64u);                                      // tile_dim1 = 64
  g1[5] = (int)(unsigned)(row_stride_elems & 0xFFFFFFFFull); // dim0_stride lo
  g1[6] = (int)(unsigned)((row_stride_elems >> 32) & 0xFFFFull);
  g1[7] = 0;
  const v4i z4 = {0, 0, 0, 0};
  const v8i z8 = {0, 0, 0, 0, 0, 0, 0, 0};
  __builtin_amdgcn_tensor_load_to_lds(g0, g1, z4, z4, z8, 0);
}

// ---------------- Kernel 1: LayerNorm (fp32 in, bf16 out) ----------------
__global__ void __launch_bounds__(256) k_ln(const float* __restrict__ x,
                                            const float* __restrict__ gamma,
                                            const float* __restrict__ beta,
                                            bf16_t* __restrict__ xn) {
  const int row = blockIdx.x;
  const int tid = threadIdx.x;
  const float4 v = ((const float4*)(x + (size_t)row * DIMC))[tid];
  __shared__ float red[16];

  float s = v.x + v.y + v.z + v.w;
  #pragma unroll
  for (int m = 1; m <= 16; m <<= 1) s += __shfl_xor(s, m, 32);
  if ((tid & 31) == 0) red[tid >> 5] = s;
  __syncthreads();
  float mean = 0.f;
  #pragma unroll
  for (int i = 0; i < 8; ++i) mean += red[i];
  mean *= (1.0f / DIMC);

  const float dx = v.x - mean, dy = v.y - mean, dz = v.z - mean, dw = v.w - mean;
  float sq = dx * dx + dy * dy + dz * dz + dw * dw;
  #pragma unroll
  for (int m = 1; m <= 16; m <<= 1) sq += __shfl_xor(sq, m, 32);
  if ((tid & 31) == 0) red[8 + (tid >> 5)] = sq;
  __syncthreads();
  float var = 0.f;
  #pragma unroll
  for (int i = 0; i < 8; ++i) var += red[8 + i];
  var *= (1.0f / DIMC);
  const float rstd = rsqrtf(var + 1e-5f);

  const int c = tid * 4;
  v4bf o;
  o[0] = (bf16_t)(dx * rstd * gamma[c + 0] + beta[c + 0]);
  o[1] = (bf16_t)(dy * rstd * gamma[c + 1] + beta[c + 1]);
  o[2] = (bf16_t)(dz * rstd * gamma[c + 2] + beta[c + 2]);
  o[3] = (bf16_t)(dw * rstd * gamma[c + 3] + beta[c + 3]);
  *(v4bf*)(xn + (size_t)row * DIMC + c) = o;
}

// ---------------- Kernel 2: fp32 -> bf16 convert ----------------
__global__ void __launch_bounds__(256) k_cvt(const float* __restrict__ src,
                                             bf16_t* __restrict__ dst, int n4) {
  const int i = blockIdx.x * blockDim.x + threadIdx.x;
  if (i < n4) {
    const float4 f = ((const float4*)src)[i];
    v4bf o;
    o[0] = (bf16_t)f.x; o[1] = (bf16_t)f.y; o[2] = (bf16_t)f.z; o[3] = (bf16_t)f.w;
    ((v4bf*)dst)[i] = o;
  }
}

// ---------------- Kernel 3: fused QKV GEMM  y = xn @ Wcat^T ----------------
__global__ void __launch_bounds__(256) k_qkv(const bf16_t* __restrict__ xn,
                                             const bf16_t* __restrict__ Wcat,
                                             bf16_t* __restrict__ q_ws,
                                             bf16_t* __restrict__ k_ws,
                                             bf16_t* __restrict__ v_ws) {
  const int tid = threadIdx.x;
  const int lane = tid & 31;
  const int w = tid >> 5;                       // 8 waves: 4(M) x 2(N)
  const int rows0 = blockIdx.x * 128 + (w & 3) * 32;
  const int cols0 = blockIdx.y * 64 + (w >> 2) * 32;

  v8f acc[2][2] = {};
  for (int kk = 0; kk < DIMC; kk += 32) {
    if (kk + 32 < DIMC) {
      __builtin_prefetch(xn + (size_t)(rows0 + lane) * DIMC + kk + 32, 0, 3);
      __builtin_prefetch(Wcat + (size_t)(cols0 + lane) * DIMC + kk + 32, 0, 3);
    }
    const v16bf a0 = load_a_frag(xn, DIMC, rows0,      kk, lane);
    const v16bf a1 = load_a_frag(xn, DIMC, rows0 + 16, kk, lane);
    const v16bf b0 = load_b_frag(Wcat, DIMC, cols0,      kk, lane);
    const v16bf b1 = load_b_frag(Wcat, DIMC, cols0 + 16, kk, lane);
    acc[0][0] = wmma_bf16(a0, b0, acc[0][0]);
    acc[0][1] = wmma_bf16(a0, b1, acc[0][1]);
    acc[1][0] = wmma_bf16(a1, b0, acc[1][0]);
    acc[1][1] = wmma_bf16(a1, b1, acc[1][1]);
  }

  const int g = lane >> 4, nn = lane & 15;
  #pragma unroll
  for (int i = 0; i < 2; ++i)
    #pragma unroll
    for (int j = 0; j < 2; ++j) {
      const int d   = cols0 + j * 16 + nn;
      const int sec = d >> 10;          // 0=Q 1=K 2=V (uniform per block)
      const int d1  = d & (DIMC - 1);
      const int h   = d1 >> 6, dh = d1 & 63;
      #pragma unroll
      for (int v = 0; v < 8; ++v) {
        const int tok = rows0 + i * 16 + g * 8 + v;
        const int b = tok >> 11, n = tok & (SEQ - 1);
        const int bhid = b * NHEAD + h;
        const float val = acc[i][j][v];
        if (sec == 0)
          q_ws[((size_t)bhid * SEQ + n) * HDIM + dh] = (bf16_t)(val * QKSCALE);
        else if (sec == 1)
          k_ws[((size_t)bhid * SEQ + n) * HDIM + dh] = (bf16_t)val;
        else
          v_ws[((size_t)bhid * HDIM + dh) * SEQ + n] = (bf16_t)val;
      }
    }
}

// ---------------- Kernel 4: flash attention with TDM-staged K/V tiles -------
// Grid: (SEQ/128, BATCH*NHEAD). 8 waves; each wave owns 16 query rows.
// Wave 0 drives the Tensor Data Mover: K-tile [64 x 64] and V-tile [64 x 64]
// per key block, double-buffered in LDS with the TDM inserting the 16B row pad.
struct AttnSmem {
  bf16_t kt[2][64][LDS_LD];   // K tile: rows = key token, cols = dh
  bf16_t vt[2][64][LDS_LD];   // V tile: rows = dh, cols = key token (V^T)
  bf16_t p[8][16][LDS_LD];    // per-wave P staging
  float  pad_;
};

__global__ void __launch_bounds__(256) k_attn(const bf16_t* __restrict__ q_ws,
                                              const bf16_t* __restrict__ k_ws,
                                              const bf16_t* __restrict__ v_ws,
                                              bf16_t* __restrict__ o_ws) {
  __shared__ AttnSmem smem;
  const int tid = threadIdx.x, lane = tid & 31, w = tid >> 5;
  const int g = lane >> 4, nn = lane & 15;
  const int bh = blockIdx.y;
  const bf16_t* Q  = q_ws + (size_t)bh * SEQ * HDIM;
  const bf16_t* K  = k_ws + (size_t)bh * SEQ * HDIM;
  const bf16_t* Vt = v_ws + (size_t)bh * HDIM * SEQ;   // [64, SEQ]
  const int q0 = blockIdx.x * 128 + w * 16;

  // LDS byte offsets (flat-pointer low dword == LDS offset for AS3 objects)
  const unsigned kt_off0 = (unsigned)(size_t)(void*)&smem.kt[0][0][0];
  const unsigned kt_off1 = (unsigned)(size_t)(void*)&smem.kt[1][0][0];
  const unsigned vt_off0 = (unsigned)(size_t)(void*)&smem.vt[0][0][0];
  const unsigned vt_off1 = (unsigned)(size_t)(void*)&smem.vt[1][0][0];

  v16bf aq[2];                                   // Q fragments, resident
  aq[0] = load_a_frag(Q, HDIM, q0, 0,  lane);
  aq[1] = load_a_frag(Q, HDIM, q0, 32, lane);

  v8f o[4] = {};
  float mrow[8], lrow[8];
  #pragma unroll
  for (int v = 0; v < 8; ++v) { mrow[v] = -1e30f; lrow[v] = 0.f; }

  // Prologue: stage key block 0 into buffer 0.
  if (w == 0) {
    tdm_load_tile(kt_off0, K, HDIM);                  // K rows stride 64
    tdm_load_tile(vt_off0, Vt, SEQ);                  // V^T rows stride 2048
  }

  int buf = 0;
  for (int kb = 0; kb < SEQ; kb += 64, buf ^= 1) {
    if (w == 0) __builtin_amdgcn_s_wait_tensorcnt(0); // current tiles landed
    __syncthreads();                                  // visible to all waves;
                                                      // other buffer now idle
    if (w == 0 && kb + 64 < SEQ) {                    // prefetch next block
      const unsigned ko = buf ? kt_off0 : kt_off1;
      const unsigned vo = buf ? vt_off0 : vt_off1;
      tdm_load_tile(ko, K + (size_t)(kb + 64) * HDIM, HDIM);
      tdm_load_tile(vo, Vt + (kb + 64), SEQ);
    }
    const bf16_t* Kl = &smem.kt[buf][0][0];
    const bf16_t* Vl = &smem.vt[buf][0][0];

    // S = Q K^T  (scale pre-applied to Q)
    v8f s[4] = {};
    #pragma unroll
    for (int t = 0; t < 4; ++t) {
      s[t] = wmma_bf16(aq[0], load_b_frag(Kl, LDS_LD, t * 16, 0,  lane), s[t]);
      s[t] = wmma_bf16(aq[1], load_b_frag(Kl, LDS_LD, t * 16, 32, lane), s[t]);
    }
    // online softmax: row stats via half-wave shfl_xor reductions
    float corr[8], mnew[8], rs[8];
    #pragma unroll
    for (int v = 0; v < 8; ++v) {
      float rm = fmaxf(fmaxf(s[0][v], s[1][v]), fmaxf(s[2][v], s[3][v]));
      #pragma unroll
      for (int m = 1; m <= 8; m <<= 1) rm = fmaxf(rm, __shfl_xor(rm, m, 32));
      mnew[v] = fmaxf(mrow[v], rm);
      corr[v] = __expf(mrow[v] - mnew[v]);
      rs[v] = 0.f;
    }
    #pragma unroll
    for (int t = 0; t < 4; ++t)
      #pragma unroll
      for (int v = 0; v < 8; ++v) {
        const float p = __expf(s[t][v] - mnew[v]);
        rs[v] += p;
        smem.p[w][g * 8 + v][t * 16 + nn] = (bf16_t)p;   // C-layout -> LDS
      }
    #pragma unroll
    for (int v = 0; v < 8; ++v) {
      #pragma unroll
      for (int m = 1; m <= 8; m <<= 1) rs[v] += __shfl_xor(rs[v], m, 32);
      lrow[v] = lrow[v] * corr[v] + rs[v];
      mrow[v] = mnew[v];
    }
    // rescale running O, then O += P @ V
    #pragma unroll
    for (int t = 0; t < 4; ++t)
      #pragma unroll
      for (int v = 0; v < 8; ++v) o[t][v] *= corr[v];

    asm volatile("s_wait_dscnt 0" ::: "memory");     // P stores -> P reads
    const v16bf ap0 = load_a_frag(&smem.p[w][0][0], LDS_LD, 0, 0,  lane);
    const v16bf ap1 = load_a_frag(&smem.p[w][0][0], LDS_LD, 0, 32, lane);
    #pragma unroll
    for (int t = 0; t < 4; ++t) {
      o[t] = wmma_bf16(ap0, load_b_frag(Vl, LDS_LD, t * 16, 0,  lane), o[t]);
      o[t] = wmma_bf16(ap1, load_b_frag(Vl, LDS_LD, t * 16, 32, lane), o[t]);
    }
    __syncthreads();                                 // done reading buf
  }

  const int b = bh >> 4, h = bh & 15;
  #pragma unroll
  for (int v = 0; v < 8; ++v) {
    const float inv = 1.0f / lrow[v];
    const int q = q0 + g * 8 + v;
    #pragma unroll
    for (int t = 0; t < 4; ++t)
      o_ws[((size_t)(b * SEQ + q)) * DIMC + h * HDIM + t * 16 + nn] =
          (bf16_t)(o[t][v] * inv);
  }
}

// ---------------- Kernel 5: output projection + bias (fp32 out) -------------
__global__ void __launch_bounds__(256) k_proj(const bf16_t* __restrict__ o_ws,
                                              const bf16_t* __restrict__ Wo,
                                              const float* __restrict__ bo,
                                              float* __restrict__ out) {
  const int tid = threadIdx.x, lane = tid & 31, w = tid >> 5;
  const int rows0 = blockIdx.x * 128 + (w & 3) * 32;
  const int cols0 = blockIdx.y * 64 + (w >> 2) * 32;

  v8f acc[2][2] = {};
  for (int kk = 0; kk < DIMC; kk += 32) {
    if (kk + 32 < DIMC) {
      __builtin_prefetch(o_ws + (size_t)(rows0 + lane) * DIMC + kk + 32, 0, 3);
      __builtin_prefetch(Wo + (size_t)(cols0 + lane) * DIMC + kk + 32, 0, 3);
    }
    const v16bf a0 = load_a_frag(o_ws, DIMC, rows0,      kk, lane);
    const v16bf a1 = load_a_frag(o_ws, DIMC, rows0 + 16, kk, lane);
    const v16bf b0 = load_b_frag(Wo, DIMC, cols0,      kk, lane);
    const v16bf b1 = load_b_frag(Wo, DIMC, cols0 + 16, kk, lane);
    acc[0][0] = wmma_bf16(a0, b0, acc[0][0]);
    acc[0][1] = wmma_bf16(a0, b1, acc[0][1]);
    acc[1][0] = wmma_bf16(a1, b0, acc[1][0]);
    acc[1][1] = wmma_bf16(a1, b1, acc[1][1]);
  }

  const int g = lane >> 4, nn = lane & 15;
  #pragma unroll
  for (int j = 0; j < 2; ++j) {
    const int d = cols0 + j * 16 + nn;
    const float bias = bo[d];
    #pragma unroll
    for (int i = 0; i < 2; ++i)
      #pragma unroll
      for (int v = 0; v < 8; ++v) {
        const int m = rows0 + i * 16 + g * 8 + v;
        out[(size_t)m * DIMC + d] = acc[i][j][v] + bias;
      }
  }
}

// ---------------- Host launcher ----------------
extern "C" void kernel_launch(void* const* d_in, const int* in_sizes, int n_in,
                              void* d_out, int out_size, void* d_ws, size_t ws_size,
                              hipStream_t stream) {
  const float* x     = (const float*)d_in[0];
  const float* gamma = (const float*)d_in[1];
  const float* beta  = (const float*)d_in[2];
  const float* Wq    = (const float*)d_in[3];
  const float* Wk    = (const float*)d_in[4];
  const float* Wv    = (const float*)d_in[5];
  const float* Wo    = (const float*)d_in[6];
  const float* bo    = (const float*)d_in[7];
  float* out = (float*)d_out;

  char*  ws  = (char*)d_ws;
  size_t off = 0;
  bf16_t* xn   = (bf16_t*)(ws + off); off += (size_t)TOKENS * DIMC * 2;      // 8 MB
  bf16_t* Wcat = (bf16_t*)(ws + off); off += (size_t)3 * DIMC * DIMC * 2;    // 6 MB
  bf16_t* Wob  = (bf16_t*)(ws + off); off += (size_t)DIMC * DIMC * 2;        // 2 MB
  bf16_t* q_ws = (bf16_t*)(ws + off); off += (size_t)TOKENS * DIMC * 2;      // 8 MB
  bf16_t* k_ws = (bf16_t*)(ws + off); off += (size_t)TOKENS * DIMC * 2;      // 8 MB
  bf16_t* v_ws = (bf16_t*)(ws + off); off += (size_t)TOKENS * DIMC * 2;      // 8 MB
  bf16_t* o_ws = (bf16_t*)(ws + off); off += (size_t)TOKENS * DIMC * 2;      // 8 MB

  k_ln<<<TOKENS, 256, 0, stream>>>(x, gamma, beta, xn);

  const int n4 = DIMC * DIMC / 4;   // 262144 -> 1024 blocks
  k_cvt<<<n4 / 256, 256, 0, stream>>>(Wq, Wcat,                   n4);
  k_cvt<<<n4 / 256, 256, 0, stream>>>(Wk, Wcat + DIMC * DIMC,     n4);
  k_cvt<<<n4 / 256, 256, 0, stream>>>(Wv, Wcat + 2 * DIMC * DIMC, n4);
  k_cvt<<<n4 / 256, 256, 0, stream>>>(Wo, Wob,                    n4);

  k_qkv<<<dim3(TOKENS / 128, 3 * DIMC / 64), 256, 0, stream>>>(xn, Wcat, q_ws, k_ws, v_ws);
  k_attn<<<dim3(SEQ / 128, BATCH * NHEAD), 256, 0, stream>>>(q_ws, k_ws, v_ws, o_ws);
  k_proj<<<dim3(TOKENS / 128, DIMC / 64), 256, 0, stream>>>(o_ws, Wob, bo, out);
}
